// StarTransformer_24635932410339
// MI455X (gfx1250) — matrix operand order repaired and checked by reading, not compiled, not run on hardware
//
#include <hip/hip_runtime.h>
#include <hip/hip_bf16.h>
#include <cstdint>

// ---------------- problem constants ----------------
#define Bc   4
#define Lc   2048
#define Hc   1024
#define Dc   1024
#define NHc  16
#define HDc  64
#define ITERSc 3
#define BLc  (Bc * Lc)          // 8192 rows
#define BLHc ((size_t)BLc * Hc) // 8,388,608 elems

typedef __bf16 bf16_t;
typedef __attribute__((ext_vector_type(16))) __bf16 v16bf;
typedef __attribute__((ext_vector_type(8)))  __bf16 bf16x8;
typedef __attribute__((ext_vector_type(8)))  float  v8f;

// ======================================================================
// fp32 -> bf16 conversion (grid-stride)
// ======================================================================
__global__ void cvt_f32_to_bf16(const float* __restrict__ src,
                                bf16_t* __restrict__ dst, size_t n) {
  size_t stride = (size_t)gridDim.x * blockDim.x;
  for (size_t i = (size_t)blockIdx.x * blockDim.x + threadIdx.x; i < n; i += stride)
    dst[i] = (bf16_t)src[i];
}

// ======================================================================
// relay init: mean over L.  thread per (b,h)
// ======================================================================
__global__ void relay_mean_kernel(const float* __restrict__ data,
                                  float* __restrict__ relay) {
  int i = blockIdx.x * blockDim.x + threadIdx.x;
  if (i >= Bc * Hc) return;
  int b = i / Hc, h = i % Hc;
  const float* p = data + (size_t)b * Lc * Hc + h;
  float s = 0.f;
  for (int l = 0; l < Lc; ++l) s += p[(size_t)l * Hc];
  relay[i] = s * (1.0f / Lc);
}

// ======================================================================
// LayerNorm, one wave32 per row, output bf16
// ======================================================================
__global__ __launch_bounds__(256)
void ln_to_bf16(const float* __restrict__ x, const float* __restrict__ scale,
                const float* __restrict__ bias, bf16_t* __restrict__ out,
                int rows) {
  int wave = blockIdx.x * (blockDim.x >> 5) + (threadIdx.x >> 5);
  int lane = threadIdx.x & 31;
  if (wave >= rows) return;
  const float* rx = x + (size_t)wave * Hc;
  float s = 0.f, ss = 0.f;
  for (int h = lane; h < Hc; h += 32) { float v = rx[h]; s += v; ss += v * v; }
  #pragma unroll
  for (int o = 16; o > 0; o >>= 1) {
    s  += __shfl_xor(s,  o, 32);
    ss += __shfl_xor(ss, o, 32);
  }
  float mu  = s * (1.0f / Hc);
  float var = ss * (1.0f / Hc) - mu * mu;
  float inv = rsqrtf(var + 1e-6f);
  bf16_t* ro = out + (size_t)wave * Hc;
  for (int h = lane; h < Hc; h += 32)
    ro[h] = (bf16_t)((rx[h] - mu) * inv * scale[h] + bias[h]);
}

// ======================================================================
// async global->LDS copy of one 32-byte slice (two b128, IOFFSET applies
// to both LDS and global address per CDNA5 ISA 10.x async pseudocode)
// ======================================================================
__device__ __forceinline__ void async_cp32(unsigned lds_off, const void* gptr) {
  asm volatile(
      "global_load_async_to_lds_b128 %0, %1, off\n\t"
      "global_load_async_to_lds_b128 %0, %1, off offset:16"
      :: "v"(lds_off), "v"(gptr) : "memory");
}

// ======================================================================
// bf16 WMMA GEMM:  C[M,N] = act( A[M,K] * Wt[N,K]^T + bias[N] )
//   block tile 128x128, 256 threads = 8 waves (4x2), wave tile 32x64,
//   K staged 32 deep in double-buffered LDS via async-to-LDS copies,
//   v_wmma_f32_16x16x32_bf16
// ======================================================================
#define LDSROW 40   // 32 + 8 pad (bf16 elems), 80B rows -> 16B aligned

template <int ACT, bool HAS_C16>
__global__ __launch_bounds__(256)
void gemm_bf16_wmma(const bf16_t* __restrict__ A,   // M x K
                    const bf16_t* __restrict__ Wt,  // N x K
                    const float*  __restrict__ bias,
                    float* __restrict__ C,
                    bf16_t* __restrict__ C16,
                    int M, int N, int K) {
  __shared__ bf16_t lA[2][128 * LDSROW];
  __shared__ bf16_t lB[2][128 * LDSROW];

  const int tid   = threadIdx.x;
  const int lane  = tid & 31;
  const int wave  = tid >> 5;
  const int waveM = wave >> 1;          // 0..3
  const int waveN = wave & 1;           // 0..1
  const int bm    = blockIdx.x * 128;
  const int bn    = blockIdx.y * 128;

  v8f acc[2][4];
  #pragma unroll
  for (int s = 0; s < 2; ++s)
    #pragma unroll
    for (int t = 0; t < 4; ++t)
      acc[s][t] = (v8f)(0.f);

  const int srow  = tid >> 1;           // 0..127
  const int scol  = (tid & 1) * 16;     // 0 / 16
  const int mrow  = lane & 15;
  const int khalf = lane >> 4;          // 0 / 1

  const bf16_t* gA = A  + (size_t)(bm + srow) * K + scol;
  const bf16_t* gB = Wt + (size_t)(bn + srow) * K + scol;
  const unsigned laOff[2] = {
      (unsigned)(uintptr_t)&lA[0][srow * LDSROW + scol],
      (unsigned)(uintptr_t)&lA[1][srow * LDSROW + scol]};
  const unsigned lbOff[2] = {
      (unsigned)(uintptr_t)&lB[0][srow * LDSROW + scol],
      (unsigned)(uintptr_t)&lB[1][srow * LDSROW + scol]};

  const int nsteps = K >> 5;
  // prologue: stage 0
  async_cp32(laOff[0], gA);
  async_cp32(lbOff[0], gB);

  for (int step = 0; step < nsteps; ++step) {
    const int cur = step & 1;
    if (step + 1 < nsteps) {
      // issue next stage, then wait for current (async loads complete
      // in order -> <=4 outstanding means stage `cur` has landed)
      async_cp32(laOff[cur ^ 1], gA + (step + 1) * 32);
      async_cp32(lbOff[cur ^ 1], gB + (step + 1) * 32);
      asm volatile("s_wait_asynccnt 4" ::: "memory");
    } else {
      asm volatile("s_wait_asynccnt 0" ::: "memory");
    }
    __syncthreads();

    v16bf afrag[2], bfrag[4];
    #pragma unroll
    for (int s = 0; s < 2; ++s) {
      // A 16x32 bf16 fragment: lanes 0-15 K{0..7,16..23}, lanes 16-31 K{8..15,24..31}
      const bf16_t* p = &lA[cur][(waveM * 32 + s * 16 + mrow) * LDSROW + khalf * 8];
      union { bf16x8 h[2]; v16bf v; } u;
      u.h[0] = *(const bf16x8*)p;
      u.h[1] = *(const bf16x8*)(p + 16);
      afrag[s] = u.v;
    }
    #pragma unroll
    for (int t = 0; t < 4; ++t) {
      // B 32x16 bf16 fragment: lanes 0-15 K{0..7,16..23}, lanes 16-31 K{8..15,24..31}
      const bf16_t* p = &lB[cur][(waveN * 64 + t * 16 + mrow) * LDSROW + khalf * 8];
      union { bf16x8 h[2]; v16bf v; } u;
      u.h[0] = *(const bf16x8*)p;
      u.h[1] = *(const bf16x8*)(p + 16);
      bfrag[t] = u.v;
    }
    #pragma unroll
    for (int s = 0; s < 2; ++s)
      #pragma unroll
      for (int t = 0; t < 4; ++t)
        acc[s][t] = __builtin_amdgcn_wmma_f32_16x16x32_bf16(
            false, afrag[s], false, bfrag[t], (short)0, acc[s][t], false, false);
    __syncthreads();   // all waves done reading buf `cur` before it is re-filled
  }

  // epilogue: VGPR j -> M=j (lanes 0-15) / M=j+8 (lanes 16-31), N=lane&15
  const int moff = khalf * 8;
  const int nlan = lane & 15;
  #pragma unroll
  for (int s = 0; s < 2; ++s)
    #pragma unroll
    for (int t = 0; t < 4; ++t) {
      int col = bn + waveN * 64 + t * 16 + nlan;
      float bv = bias[col];
      #pragma unroll
      for (int j = 0; j < 8; ++j) {
        int row = bm + waveM * 32 + s * 16 + moff + j;
        float v = acc[s][t][j] + bv;
        if (ACT == 1) v = v > 0.f ? v : 0.01f * v;   // leaky_relu
        size_t idx = (size_t)row * N + col;
        C[idx] = v;
        if (HAS_C16) C16[idx] = (bf16_t)v;
      }
    }
}

// ======================================================================
// relay projections (tiny, fp32):  kr/vr (ring aux) + sq/skr/svr (star)
// ======================================================================
__global__ void relay_project(const float* __restrict__ relay,
                              const float* Wrk, const float* brk,
                              const float* Wrv, const float* brv,
                              const float* Wsq, const float* bsq,
                              const float* Wsk, const float* bsk,
                              const float* Wsv, const float* bsv,
                              float* kr, float* vr, float* sq,
                              float* skr, float* svr) {
  int i = blockIdx.x * blockDim.x + threadIdx.x;
  if (i >= 5 * Bc * Dc) return;
  int which = i / (Bc * Dc);
  int r = i % (Bc * Dc);
  int b = r / Dc, n = r % Dc;
  const float *W, *bb; float* o;
  switch (which) {
    case 0: W = Wrk; bb = brk; o = kr;  break;
    case 1: W = Wrv; bb = brv; o = vr;  break;
    case 2: W = Wsq; bb = bsq; o = sq;  break;
    case 3: W = Wsk; bb = bsk; o = skr; break;
    default: W = Wsv; bb = bsv; o = svr; break;
  }
  const float* rb = relay + (size_t)b * Hc;
  const float* wr = W + (size_t)n * Hc;
  float s = bb[n];
  for (int h = 0; h < Hc; ++h) s += rb[h] * wr[h];
  o[r] = s;
}

// ======================================================================
// ring (windowed) attention: one wave per (b,l,head), 2 dims per lane
// keys = {k[l-1], k[l], k[l+1] (zero padded), ke[l], k_relay}
// ======================================================================
__global__ __launch_bounds__(256)
void ring_attention(const float* __restrict__ q, const float* __restrict__ k,
                    const float* __restrict__ v, const float* __restrict__ ke,
                    const float* __restrict__ ve, const float* __restrict__ kr,
                    const float* __restrict__ vr, bf16_t* __restrict__ att) {
  int gw = blockIdx.x * (blockDim.x >> 5) + (threadIdx.x >> 5);
  int lane = threadIdx.x & 31;
  if (gw >= Bc * Lc * NHc) return;
  int n = gw % NHc;
  int bl = gw / NHc;           // b*L + l
  int l = bl % Lc, b = bl / Lc;
  size_t rowb = (size_t)bl * Dc + n * HDc;
  int d0 = lane * 2;
  float q0 = q[rowb + d0], q1 = q[rowb + d0 + 1];

  float sc[5];
  #pragma unroll
  for (int w = 0; w < 3; ++w) {
    int ll = l + w - 1;
    float p = 0.f;
    if (ll >= 0 && ll < Lc) {
      size_t kb = ((size_t)b * Lc + ll) * Dc + n * HDc + d0;
      p = q0 * k[kb] + q1 * k[kb + 1];
    }
    sc[w] = p;
  }
  sc[3] = q0 * ke[rowb + d0] + q1 * ke[rowb + d0 + 1];
  {
    size_t kb = (size_t)b * Dc + n * HDc + d0;
    sc[4] = q0 * kr[kb] + q1 * kr[kb + 1];
  }
  #pragma unroll
  for (int w = 0; w < 5; ++w) {
    #pragma unroll
    for (int o = 16; o > 0; o >>= 1) sc[w] += __shfl_xor(sc[w], o, 32);
    sc[w] *= 0.125f;  // 1/sqrt(64)
  }
  float mx = sc[0];
  #pragma unroll
  for (int w = 1; w < 5; ++w) mx = fmaxf(mx, sc[w]);
  float e[5], sum = 0.f;
  #pragma unroll
  for (int w = 0; w < 5; ++w) { e[w] = __expf(sc[w] - mx); sum += e[w]; }
  float inv = 1.f / sum;
  float o0 = 0.f, o1 = 0.f;
  #pragma unroll
  for (int w = 0; w < 3; ++w) {
    int ll = l + w - 1;
    if (ll >= 0 && ll < Lc) {
      size_t vb = ((size_t)b * Lc + ll) * Dc + n * HDc + d0;
      float a = e[w] * inv;
      o0 += a * v[vb]; o1 += a * v[vb + 1];
    }
  }
  { float a = e[3] * inv; o0 += a * ve[rowb + d0]; o1 += a * ve[rowb + d0 + 1]; }
  { size_t vb = (size_t)b * Dc + n * HDc + d0;
    float a = e[4] * inv; o0 += a * vr[vb]; o1 += a * vr[vb + 1]; }
  att[rowb + d0]     = (bf16_t)o0;
  att[rowb + d0 + 1] = (bf16_t)o1;
}

// ======================================================================
// star attention: one block per (b,head); softmax over L+1 keys in LDS
// ======================================================================
__global__ __launch_bounds__(256)
void star_attention(const float* __restrict__ sq,
                    const float* __restrict__ skn, const float* __restrict__ svn,
                    const float* __restrict__ skr, const float* __restrict__ svr,
                    const int* __restrict__ mask, float* __restrict__ satt) {
  __shared__ float alpha[Lc + 1];
  __shared__ float red[256];
  __shared__ float qs[HDc];
  int bh = blockIdx.x;
  int b = bh / NHc, n = bh % NHc;
  int t = threadIdx.x;
  if (t < HDc) qs[t] = sq[(size_t)b * Dc + n * HDc + t];
  __syncthreads();

  float lmax = -3.0e30f;
  for (int j = t; j <= Lc; j += 256) {
    float s;
    if (j == 0) {
      const float* kp = skr + (size_t)b * Dc + n * HDc;
      s = 0.f;
      for (int d = 0; d < HDc; ++d) s += qs[d] * kp[d];
      s *= 0.125f;
    } else {
      int l = j - 1;
      if (mask[b * Lc + l] == 0) s = -1.0e30f;
      else {
        const float* kp = skn + ((size_t)b * Lc + l) * Dc + n * HDc;
        s = 0.f;
        for (int d = 0; d < HDc; ++d) s += qs[d] * kp[d];
        s *= 0.125f;
      }
    }
    alpha[j] = s;
    lmax = fmaxf(lmax, s);
  }
  red[t] = lmax; __syncthreads();
  for (int o = 128; o > 0; o >>= 1) {
    if (t < o) red[t] = fmaxf(red[t], red[t + o]);
    __syncthreads();
  }
  float mx = red[0]; __syncthreads();

  float lsum = 0.f;
  for (int j = t; j <= Lc; j += 256) {
    float e = __expf(alpha[j] - mx);
    alpha[j] = e; lsum += e;
  }
  __syncthreads();
  red[t] = lsum; __syncthreads();
  for (int o = 128; o > 0; o >>= 1) {
    if (t < o) red[t] += red[t + o];
    __syncthreads();
  }
  float inv = 1.f / red[0];
  __syncthreads();

  if (t < HDc) {
    int d = t;
    float o = alpha[0] * inv * svr[(size_t)b * Dc + n * HDc + d];
    for (int j = 1; j <= Lc; ++j)
      o += alpha[j] * inv * svn[((size_t)b * Lc + j - 1) * Dc + n * HDc + d];
    satt[(size_t)b * Dc + n * HDc + d] = o;
  }
}

// ======================================================================
// star output projection (tiny): relay = leaky(satt @ Wo^T + b)
// ======================================================================
__global__ void star_out_kernel(const float* __restrict__ satt,
                                const float* __restrict__ W,
                                const float* __restrict__ bvec,
                                float* __restrict__ relay) {
  int i = blockIdx.x * blockDim.x + threadIdx.x;
  if (i >= Bc * Hc) return;
  int b = i / Hc, h = i % Hc;
  const float* sp = satt + (size_t)b * Dc;
  const float* wp = W + (size_t)h * Dc;
  float s = bvec[h];
  for (int d = 0; d < Dc; ++d) s += sp[d] * wp[d];
  relay[i] = s > 0.f ? s : 0.01f * s;
}

// ======================================================================
// zero nodes at padded positions
// ======================================================================
__global__ void mask_nodes_kernel(float* __restrict__ nodes,
                                  const int* __restrict__ mask) {
  size_t stride = (size_t)gridDim.x * blockDim.x;
  for (size_t i = (size_t)blockIdx.x * blockDim.x + threadIdx.x; i < BLHc; i += stride) {
    int bl = (int)(i / Hc);
    if (mask[bl] == 0) nodes[i] = 0.f;
  }
}

// ======================================================================
// final output copy: (nodes flat, relay flat)
// ======================================================================
__global__ void write_out_kernel(const float* __restrict__ nodes,
                                 const float* __restrict__ relay,
                                 float* __restrict__ out) {
  size_t total = BLHc + (size_t)Bc * Hc;
  size_t stride = (size_t)gridDim.x * blockDim.x;
  for (size_t i = (size_t)blockIdx.x * blockDim.x + threadIdx.x; i < total; i += stride)
    out[i] = (i < BLHc) ? nodes[i] : relay[i - BLHc];
}

// ======================================================================
// host side
// ======================================================================
extern "C" void kernel_launch(void* const* d_in, const int* in_sizes, int n_in,
                              void* d_out, int out_size, void* d_ws, size_t ws_size,
                              hipStream_t stream) {
  const float* data       = (const float*)d_in[0];
  const int*   mask       = (const int*)d_in[1];
  const float* ln_scale   = (const float*)d_in[2];
  const float* ln_bias    = (const float*)d_in[3];
  const float* ring_qkv_w = (const float*)d_in[4];
  const float* ring_qkv_b = (const float*)d_in[5];
  const float* ring_o_w   = (const float*)d_in[6];
  const float* ring_o_b   = (const float*)d_in[7];
  const float* star_qkv_w = (const float*)d_in[8];
  const float* star_qkv_b = (const float*)d_in[9];
  const float* star_o_w   = (const float*)d_in[10];
  const float* star_o_b   = (const float*)d_in[11];
  float* out = (float*)d_out;

  char* ws = (char*)d_ws;
  size_t off = 0;
  auto alloc = [&](size_t bytes) -> void* {
    void* p = ws + off;
    off = (off + bytes + 255) & ~(size_t)255;
    return p;
  };

  const size_t WMAT = (size_t)Dc * Hc;       // 1M elems per weight matrix
  bf16_t* wb_rqkv = (bf16_t*)alloc(9 * WMAT * 2);
  bf16_t* wb_ro   = (bf16_t*)alloc(3 * WMAT * 2);
  bf16_t* wb_sqkv = (bf16_t*)alloc(9 * WMAT * 2);
  bf16_t* embs16  = (bf16_t*)alloc(BLHc * 2);
  bf16_t* xn16    = (bf16_t*)alloc(BLHc * 2);
  bf16_t* att16   = (bf16_t*)alloc(BLHc * 2);
  bf16_t* y16     = (bf16_t*)alloc(BLHc * 2);
  float* qb    = (float*)alloc(BLHc * 4);
  float* kb    = (float*)alloc(BLHc * 4);
  float* vb    = (float*)alloc(BLHc * 4);
  float* keb   = (float*)alloc(BLHc * 4);
  float* veb   = (float*)alloc(BLHc * 4);
  float* nodes = (float*)alloc(BLHc * 4);
  float* relay = (float*)alloc((size_t)Bc * Hc * 4);
  float* krb   = (float*)alloc((size_t)Bc * Dc * 4);
  float* vrb   = (float*)alloc((size_t)Bc * Dc * 4);
  float* sqb   = (float*)alloc((size_t)Bc * Dc * 4);
  float* skrb  = (float*)alloc((size_t)Bc * Dc * 4);
  float* svrb  = (float*)alloc((size_t)Bc * Dc * 4);
  float* sattb = (float*)alloc((size_t)Bc * Dc * 4);
  // alias star K/V node projections onto q/k buffers (dead after ring attn)
  float* sknb = qb;
  float* svnb = kb;

  // ---- one-time conversions ----
  cvt_f32_to_bf16<<<1024, 256, 0, stream>>>(ring_qkv_w, wb_rqkv, 9 * WMAT);
  cvt_f32_to_bf16<<<1024, 256, 0, stream>>>(ring_o_w,   wb_ro,   3 * WMAT);
  cvt_f32_to_bf16<<<1024, 256, 0, stream>>>(star_qkv_w, wb_sqkv, 9 * WMAT);
  cvt_f32_to_bf16<<<2048, 256, 0, stream>>>(data, embs16, BLHc);
  relay_mean_kernel<<<(Bc * Hc + 255) / 256, 256, 0, stream>>>(data, relay);

  dim3 ggrid(BLc / 128, Dc / 128);   // 64 x 8
  for (int i = 0; i < ITERSc; ++i) {
    const float* src = (i == 0) ? data : nodes;
    // 1) layernorm -> bf16
    ln_to_bf16<<<BLc / 8, 256, 0, stream>>>(src, ln_scale + i * Hc,
                                            ln_bias + i * Hc, xn16, BLc);
    // 2-4) ring q/k/v GEMMs
    gemm_bf16_wmma<0, false><<<ggrid, 256, 0, stream>>>(
        xn16, wb_rqkv + (size_t)(i * 3 + 0) * WMAT,
        ring_qkv_b + (size_t)(i * 3 + 0) * Dc, qb, nullptr, BLc, Dc, Hc);
    gemm_bf16_wmma<0, false><<<ggrid, 256, 0, stream>>>(
        xn16, wb_rqkv + (size_t)(i * 3 + 1) * WMAT,
        ring_qkv_b + (size_t)(i * 3 + 1) * Dc, kb, nullptr, BLc, Dc, Hc);
    gemm_bf16_wmma<0, false><<<ggrid, 256, 0, stream>>>(
        xn16, wb_rqkv + (size_t)(i * 3 + 2) * WMAT,
        ring_qkv_b + (size_t)(i * 3 + 2) * Dc, vb, nullptr, BLc, Dc, Hc);
    // 5) aux K/V on original embeddings
    gemm_bf16_wmma<0, false><<<ggrid, 256, 0, stream>>>(
        embs16, wb_rqkv + (size_t)(i * 3 + 1) * WMAT,
        ring_qkv_b + (size_t)(i * 3 + 1) * Dc, keb, nullptr, BLc, Dc, Hc);
    gemm_bf16_wmma<0, false><<<ggrid, 256, 0, stream>>>(
        embs16, wb_rqkv + (size_t)(i * 3 + 2) * WMAT,
        ring_qkv_b + (size_t)(i * 3 + 2) * Dc, veb, nullptr, BLc, Dc, Hc);
    // 6) relay projections (tiny, fp32 weights)
    relay_project<<<(5 * Bc * Dc + 255) / 256, 256, 0, stream>>>(relay,
        ring_qkv_w + (size_t)(i * 3 + 1) * WMAT, ring_qkv_b + (size_t)(i * 3 + 1) * Dc,
        ring_qkv_w + (size_t)(i * 3 + 2) * WMAT, ring_qkv_b + (size_t)(i * 3 + 2) * Dc,
        star_qkv_w + (size_t)(i * 3 + 0) * WMAT, star_qkv_b + (size_t)(i * 3 + 0) * Dc,
        star_qkv_w + (size_t)(i * 3 + 1) * WMAT, star_qkv_b + (size_t)(i * 3 + 1) * Dc,
        star_qkv_w + (size_t)(i * 3 + 2) * WMAT, star_qkv_b + (size_t)(i * 3 + 2) * Dc,
        krb, vrb, sqb, skrb, svrb);
    // 7) windowed ring attention -> att (bf16)
    ring_attention<<<(Bc * Lc * NHc) / 8, 256, 0, stream>>>(qb, kb, vb, keb, veb,
                                                            krb, vrb, att16);
    // 8) ring output projection + leaky_relu -> nodes (fp32) + y (bf16)
    gemm_bf16_wmma<1, true><<<ggrid, 256, 0, stream>>>(
        att16, wb_ro + (size_t)i * WMAT,
        ring_o_b + (size_t)i * Hc, nodes, y16, BLc, Hc, Dc);
    // 9) star K/V over nodes (relay row handled by relay_project)
    gemm_bf16_wmma<0, false><<<ggrid, 256, 0, stream>>>(
        y16, wb_sqkv + (size_t)(i * 3 + 1) * WMAT,
        star_qkv_b + (size_t)(i * 3 + 1) * Dc, sknb, nullptr, BLc, Dc, Hc);
    gemm_bf16_wmma<0, false><<<ggrid, 256, 0, stream>>>(
        y16, wb_sqkv + (size_t)(i * 3 + 2) * WMAT,
        star_qkv_b + (size_t)(i * 3 + 2) * Dc, svnb, nullptr, BLc, Dc, Hc);
    // 10) star attention + output
    star_attention<<<Bc * NHc, 256, 0, stream>>>(sqb, sknb, svnb, skrb, svrb, mask, sattb);
    star_out_kernel<<<(Bc * Hc + 255) / 256, 256, 0, stream>>>(sattb,
        star_o_w + (size_t)i * WMAT, star_o_b + (size_t)i * Hc, relay);
    // 11) zero padded nodes
    mask_nodes_kernel<<<4096, 256, 0, stream>>>(nodes, mask);
  }
  write_out_kernel<<<8192, 256, 0, stream>>>(nodes, relay, out);
  (void)in_sizes; (void)n_in; (void)out_size; (void)ws_size;
}
